// CL_VAE_89094801588751
// MI455X (gfx1250) — matrix operand order repaired and checked by reading, not compiled, not run on hardware
//
#include <hip/hip_runtime.h>

typedef __attribute__((ext_vector_type(16))) _Float16 v16h;
typedef __attribute__((ext_vector_type(8)))  _Float16 v8h;
typedef __attribute__((ext_vector_type(8)))  float    v8f;

#define B_DIM 1024
#define I_DIM 20000
#define D_DIM 200
#define U_DIM 512
#define L_DIM 200
#define EPSF  1e-8f

#define DP1 208    // h_pre padded cols (13 * 16)
#define DP2 224    // h f16 padded K for GEMM2 (7 * 32)
#define LDA 56     // LDS tile row stride in halves: 112B = 28-bank stride,
                   // gcd(28,64)=4 -> 16 conflict-free start banks {0,4,...,60}
#define KSPLIT 25  // GEMM1 K splits (25 k-steps of 32 each = 800 K per split)

// workspace layout (bytes)
#define OFF_HPART  0u                 // 25 * 1024*208*4 = 21299200 (non-atomic partials)
#define OFF_ACC    21299200u          // 2 floats (acc[0]=base sum, acc[1]=kl sum)
#define OFF_LSE    21299328u          // 1024 floats
#define OFF_H16    21303424u          // 1024*224*2 = 458752
#define OFF_LOGITS 21762304u          // 1024*20000*4 = 81920000 (~103.7MB total)

#define HPART_STRIDE ((size_t)B_DIM * DP1)

union F16Frag { v16h v; v8h h[2]; };

__device__ __forceinline__ v8h pack8(float4 a, float4 b) {
  v8h p;
  p[0] = (_Float16)a.x; p[1] = (_Float16)a.y; p[2] = (_Float16)a.z; p[3] = (_Float16)a.w;
  p[4] = (_Float16)b.x; p[5] = (_Float16)b.y; p[6] = (_Float16)b.z; p[7] = (_Float16)b.w;
  return p;
}
__device__ __forceinline__ v8h zero8() {
  v8h z = {(_Float16)0.f,(_Float16)0.f,(_Float16)0.f,(_Float16)0.f,
           (_Float16)0.f,(_Float16)0.f,(_Float16)0.f,(_Float16)0.f};
  return z;
}
__device__ __forceinline__ float wred_sum(float v) {
  for (int o = 16; o > 0; o >>= 1) v += __shfl_xor(v, o, 32);
  return v;
}
__device__ __forceinline__ float wred_max(float v) {
  for (int o = 16; o > 0; o >>= 1) v = fmaxf(v, __shfl_xor(v, o, 32));
  return v;
}

__global__ void k_zero(float* __restrict__ p, int n) {
  int i = blockIdx.x * blockDim.x + threadIdx.x;
  if (i < n) p[i] = 0.f;
}

// ---------------------------------------------------------------------------
// GEMM1: hpart[ksplit] = rating[1024 x 20000] * enc_w^T -> [1024 x 208]
// grid (8, 25): x = M-block of 128, y = K-split. Non-atomic partial outputs.
// ---------------------------------------------------------------------------
__device__ __forceinline__ void stage1(const float* __restrict__ rating,
                                       const float* __restrict__ encw,
                                       _Float16* __restrict__ As,
                                       _Float16* __restrict__ Bs,
                                       int m_blk, int kg, int tid, bool pf) {
  // A tile 128x32 (f32 -> f16): 256 tasks = (row, 16-col half)
  {
    const int r = tid >> 1, hh = tid & 1;
    const float* gp = rating + (size_t)(m_blk + r) * I_DIM + kg + hh * 16;
    const float4* g = (const float4*)gp;
    if (pf) __builtin_prefetch(gp + 64, 0, 0);
    _Float16* dst = As + r * LDA + hh * 16;
    *(v8h*)dst       = pack8(g[0], g[1]);
    *(v8h*)(dst + 8) = pack8(g[2], g[3]);
  }
  // B tile 208x32 from enc_w[200 x 20000], rows >= 200 -> 0
  for (int task = tid; task < 416; task += 256) {
    const int r = task >> 1, c = task & 1;
    _Float16* dst = Bs + r * LDA + c * 16;
    if (r < D_DIM) {
      const float4* g = (const float4*)(encw + (size_t)r * I_DIM + kg + c * 16);
      *(v8h*)dst       = pack8(g[0], g[1]);
      *(v8h*)(dst + 8) = pack8(g[2], g[3]);
    } else {
      *(v8h*)dst = zero8(); *(v8h*)(dst + 8) = zero8();
    }
  }
}

__global__ __launch_bounds__(256)
void k_gemm1(const float* __restrict__ rating, const float* __restrict__ encw,
             float* __restrict__ hpart) {
  __shared__ _Float16 As[2][128 * LDA] __attribute__((aligned(16)));
  __shared__ _Float16 Bs[2][208 * LDA] __attribute__((aligned(16)));
  const int tid  = threadIdx.x;
  const int lane = tid & 31, wid = tid >> 5;
  const int m_blk = blockIdx.x * 128;
  const int kbase = blockIdx.y * 800;

  v8f acc[13];
  v8f zf = {0.f,0.f,0.f,0.f,0.f,0.f,0.f,0.f};
#pragma unroll
  for (int i = 0; i < 13; ++i) acc[i] = zf;

  stage1(rating, encw, As[0], Bs[0], m_blk, kbase, tid, true);

  for (int kk = 0; kk < 25; ++kk) {
    __syncthreads();
    const int cur = kk & 1;
    if (kk < 24)
      stage1(rating, encw, As[cur ^ 1], Bs[cur ^ 1], m_blk,
             kbase + (kk + 1) * 32, tid, kk < 23);

    // A fragment: lane L holds row m0+(L&15), K (L>>4)*16..+15
    F16Frag a;
    {
      const _Float16* ap = As[cur] + (wid * 16 + (lane & 15)) * LDA + (lane >> 4) * 16;
      a.h[0] = *(const v8h*)ap;
      a.h[1] = *(const v8h*)(ap + 8);
    }
    // 3-deep B-fragment pipeline; sched_barrier pins loads ahead of wmma so
    // the backend keeps distinct regs and emits partial s_wait_dscnt.
    const _Float16* bbase = Bs[cur] + (lane & 15) * LDA + (lane >> 4) * 16;
    F16Frag bb[3];
    bb[0].h[0] = *(const v8h*)bbase;
    bb[0].h[1] = *(const v8h*)(bbase + 8);
    bb[1].h[0] = *(const v8h*)(bbase + 16 * LDA);
    bb[1].h[1] = *(const v8h*)(bbase + 16 * LDA + 8);
#pragma unroll
    for (int nt = 0; nt < 13; ++nt) {
      if (nt + 2 < 13) {
        const _Float16* bp = bbase + (nt + 2) * 16 * LDA;
        bb[(nt + 2) % 3].h[0] = *(const v8h*)bp;
        bb[(nt + 2) % 3].h[1] = *(const v8h*)(bp + 8);
      }
      __builtin_amdgcn_sched_barrier(0);
      acc[nt] = __builtin_amdgcn_wmma_f32_16x16x32_f16(
          false, a.v, false, bb[nt % 3].v, (short)0, acc[nt], false, false);
    }
  }

  // C layout: VGPR r -> M = r (lanes 0-15) or 8+r (lanes 16-31); N = lane&15
  float* hp = hpart + (size_t)blockIdx.y * HPART_STRIDE;
  const int m0 = m_blk + wid * 16 + (lane >> 4) * 8;
#pragma unroll
  for (int nt = 0; nt < 13; ++nt) {
    const int n = nt * 16 + (lane & 15);
#pragma unroll
    for (int r = 0; r < 8; ++r)
      hp[(size_t)(m0 + r) * DP1 + n] = acc[nt][r];
  }
}

// ---------------------------------------------------------------------------
// h16[m][d] = tanh(sum_k hpart[k] + enc_b), padded to DP2=224 cols, f16
// ---------------------------------------------------------------------------
__global__ __launch_bounds__(256)
void k_act(const float* __restrict__ hpart, const float* __restrict__ encb,
           _Float16* __restrict__ h16) {
  const int idx = blockIdx.x * 256 + threadIdx.x;
  if (idx >= B_DIM * DP2) return;
  const int m = idx / DP2, d = idx % DP2;
  float v = 0.f;
  if (d < D_DIM) {
    float s = 0.f;
    const float* p = hpart + (size_t)m * DP1 + d;
#pragma unroll
    for (int k = 0; k < KSPLIT; ++k) s += p[k * HPART_STRIDE];
    v = tanhf(s + encb[d]);
  }
  h16[idx] = (_Float16)v;
}

// ---------------------------------------------------------------------------
// GEMM2: logits[1024 x 20000] = h16[1024 x 224] * dec_w^T + dec_b
// grid (157, 8): x = N-block 128, y = M-block 128; K = 224 = 7 steps of 32
// ---------------------------------------------------------------------------
__device__ __forceinline__ void stage2(const float* __restrict__ decw,
                                       _Float16* __restrict__ Bs,
                                       int n_blk, int kb, int tid) {
  const int r = tid >> 1, c = tid & 1;
  const int n = n_blk + r;
  const int base = kb + c * 16;
  _Float16* dst = Bs + r * LDA + c * 16;
  if (n < I_DIM && base + 16 <= D_DIM) {
    const float4* g = (const float4*)(decw + (size_t)n * D_DIM + base);
    *(v8h*)dst       = pack8(g[0], g[1]);
    *(v8h*)(dst + 8) = pack8(g[2], g[3]);
  } else {
    v8h p0 = zero8(), p1 = zero8();
    if (n < I_DIM) {
#pragma unroll
      for (int j = 0; j < 16; ++j) {
        const int col = base + j;
        const float v = (col < D_DIM) ? decw[(size_t)n * D_DIM + col] : 0.f;
        if (j < 8) p0[j] = (_Float16)v; else p1[j - 8] = (_Float16)v;
      }
    }
    *(v8h*)dst = p0; *(v8h*)(dst + 8) = p1;
  }
}

__global__ __launch_bounds__(256)
void k_gemm2(const _Float16* __restrict__ h16, const float* __restrict__ decw,
             const float* __restrict__ decb, float* __restrict__ logits) {
  __shared__ _Float16 Bs[2][128 * LDA] __attribute__((aligned(16)));
  const int tid  = threadIdx.x;
  const int lane = tid & 31, wid = tid >> 5;
  const int n_blk = blockIdx.x * 128;
  const int m_blk = blockIdx.y * 128;
  const int m0w = m_blk + wid * 16;

  v8f acc[8];
  v8f zf = {0.f,0.f,0.f,0.f,0.f,0.f,0.f,0.f};
#pragma unroll
  for (int i = 0; i < 8; ++i) acc[i] = zf;

  stage2(decw, Bs[0], n_blk, 0, tid);

  for (int kk = 0; kk < 7; ++kk) {
    __syncthreads();
    const int cur = kk & 1;
    const int kb = kk * 32;
    if (kk < 6) stage2(decw, Bs[cur ^ 1], n_blk, kb + 32, tid);

    // A fragment straight from global h16 (L2-resident, 16B aligned chunks)
    F16Frag a;
    {
      const _Float16* ap = h16 + (size_t)(m0w + (lane & 15)) * DP2 + kb + (lane >> 4) * 16;
      a.h[0] = *(const v8h*)ap;
      a.h[1] = *(const v8h*)(ap + 8);
    }
    const _Float16* bbase = Bs[cur] + (lane & 15) * LDA + (lane >> 4) * 16;
    F16Frag bb[3];
    bb[0].h[0] = *(const v8h*)bbase;
    bb[0].h[1] = *(const v8h*)(bbase + 8);
    bb[1].h[0] = *(const v8h*)(bbase + 16 * LDA);
    bb[1].h[1] = *(const v8h*)(bbase + 16 * LDA + 8);
#pragma unroll
    for (int nt = 0; nt < 8; ++nt) {
      if (nt + 2 < 8) {
        const _Float16* bp = bbase + (nt + 2) * 16 * LDA;
        bb[(nt + 2) % 3].h[0] = *(const v8h*)bp;
        bb[(nt + 2) % 3].h[1] = *(const v8h*)(bp + 8);
      }
      __builtin_amdgcn_sched_barrier(0);
      acc[nt] = __builtin_amdgcn_wmma_f32_16x16x32_f16(
          false, a.v, false, bb[nt % 3].v, (short)0, acc[nt], false, false);
    }
  }

  const int m0 = m0w + (lane >> 4) * 8;
#pragma unroll
  for (int nt = 0; nt < 8; ++nt) {
    const int n = n_blk + nt * 16 + (lane & 15);
    if (n < I_DIM) {
      const float bias = decb[n];
#pragma unroll
      for (int r = 0; r < 8; ++r)
        logits[(size_t)(m0 + r) * I_DIM + n] = acc[nt][r] + bias;
    }
  }
}

// ---------------------------------------------------------------------------
// Per-row: cache row in LDS (80KB dynamic), compute max, sum(r), sum(logits*r),
// then lse; accumulate row loss = sum(logits*r) - lse*sum(r). One block/row.
// ---------------------------------------------------------------------------
__global__ __launch_bounds__(256)
void k_rowstats(const float* __restrict__ logits, const float* __restrict__ rating,
                float* __restrict__ lse_out, float* __restrict__ acc) {
  extern __shared__ float row[] __attribute__((aligned(16)));
  __shared__ float red[24];
  __shared__ float tot[3];
  const int b = blockIdx.x, tid = threadIdx.x;
  const int lane = tid & 31, wid = tid >> 5;
  const float4* lg4 = (const float4*)(logits + (size_t)b * I_DIM);
  const float4* rt4 = (const float4*)(rating + (size_t)b * I_DIM);
  float4* row4 = (float4*)row;

  float mx = -3.4e38f, s0 = 0.f, s1 = 0.f;
  for (int i = tid; i < I_DIM / 4; i += 256) {
    const float4 v = lg4[i];
    const float4 r = rt4[i];
    row4[i] = v;
    mx = fmaxf(mx, fmaxf(fmaxf(v.x, v.y), fmaxf(v.z, v.w)));
    s1 += v.x * r.x + v.y * r.y + v.z * r.z + v.w * r.w;
    s0 += r.x + r.y + r.z + r.w;
  }
  mx = wred_max(mx); s0 = wred_sum(s0); s1 = wred_sum(s1);
  if (lane == 0) { red[wid] = mx; red[8 + wid] = s0; red[16 + wid] = s1; }
  __syncthreads();
  if (tid == 0) {
    float M = red[0], a0 = red[8], a1 = red[16];
    for (int i = 1; i < 8; ++i) {
      M = fmaxf(M, red[i]); a0 += red[8 + i]; a1 += red[16 + i];
    }
    tot[0] = M; tot[1] = a0; tot[2] = a1;
  }
  __syncthreads();
  const float M = tot[0];
  float s2 = 0.f;
  for (int i = tid; i < I_DIM; i += 256) s2 += __expf(row[i] - M);
  s2 = wred_sum(s2);
  __syncthreads();
  if (lane == 0) red[wid] = s2;
  __syncthreads();
  if (tid == 0) {
    float S = 0.f;
    for (int i = 0; i < 8; ++i) S += red[i];
    const float lse = M + __logf(S);
    lse_out[b] = lse;
    atomicAdd(acc, tot[2] - lse * tot[1]);
  }
}

// ---------------------------------------------------------------------------
// KL: one block per user; gather logits/before at inter_idx, masked mean,
// scaled by piw, accumulated.
// ---------------------------------------------------------------------------
__global__ __launch_bounds__(256)
void k_kl(const float* __restrict__ logits, const float* __restrict__ lse_arr,
          const float* __restrict__ before, const float* __restrict__ piw,
          const int* __restrict__ bidx, const int* __restrict__ iidx,
          const int* __restrict__ lens, float* __restrict__ acc) {
  __shared__ float red[8];
  const int u = blockIdx.x, tid = threadIdx.x;
  const int lane = tid & 31, wid = tid >> 5;
  const int bu  = bidx[u];
  const int len = lens[u];
  const float lse = lse_arr[bu];
  float term = 0.f;
  if (tid < len) {
    const int i = iidx[u * L_DIM + tid];
    const float lp = (logits[(size_t)bu * I_DIM + i] - lse) + EPSF;
    const float bs = before[(size_t)u * I_DIM + i];
    const float lb = __logf(bs) + EPSF;
    term = __expf(lb) * (lb - lp);
  }
  term = wred_sum(term);
  if (lane == 0) red[wid] = term;
  __syncthreads();
  if (tid == 0) {
    float S = 0.f;
    for (int i = 0; i < 8; ++i) S += red[i];
    atomicAdd(acc + 1, piw[u] * S / (float)len);
  }
}

__global__ void k_fin(const float* __restrict__ acc, float* __restrict__ out) {
  out[0] = -acc[0] / (float)B_DIM;
  out[1] =  acc[1] / (float)U_DIM;
}

extern "C" void kernel_launch(void* const* d_in, const int* in_sizes, int n_in,
                              void* d_out, int out_size, void* d_ws, size_t ws_size,
                              hipStream_t stream) {
  (void)in_sizes; (void)n_in; (void)out_size; (void)ws_size;
  const float* rating = (const float*)d_in[0];
  const float* encw   = (const float*)d_in[1];
  const float* encb   = (const float*)d_in[2];
  const float* decw   = (const float*)d_in[3];
  const float* decb   = (const float*)d_in[4];
  const float* before = (const float*)d_in[5];
  const float* piw    = (const float*)d_in[6];
  const int*   bidx   = (const int*)d_in[7];
  const int*   iidx   = (const int*)d_in[8];
  const int*   lens   = (const int*)d_in[9];
  float* out = (float*)d_out;

  char* ws = (char*)d_ws;
  float*    hpart  = (float*)(ws + OFF_HPART);
  float*    acc    = (float*)(ws + OFF_ACC);
  float*    lsea   = (float*)(ws + OFF_LSE);
  _Float16* h16    = (_Float16*)(ws + OFF_H16);
  float*    logits = (float*)(ws + OFF_LOGITS);

  k_zero<<<1, 32, 0, stream>>>(acc, 2);
  k_gemm1<<<dim3(8, KSPLIT), 256, 0, stream>>>(rating, encw, hpart);
  k_act<<<(B_DIM * DP2) / 256, 256, 0, stream>>>(hpart, encb, h16);
  k_gemm2<<<dim3(157, 8), 256, 0, stream>>>(h16, decw, decb, logits);
  k_rowstats<<<B_DIM, 256, I_DIM * sizeof(float), stream>>>(logits, rating, lsea, acc);
  k_kl<<<U_DIM, 256, 0, stream>>>(logits, lsea, before, piw, bidx, iidx, lens, acc);
  k_fin<<<1, 1, 0, stream>>>(acc, out);
}
